// EP_GAT_PS_64493228917299
// MI455X (gfx1250) — matrix operand order repaired and checked by reading, not compiled, not run on hardware
//
#include <hip/hip_runtime.h>
#include <hip/hip_bf16.h>

// CDNA5 / gfx1250 implementation of the EP-GAT reference.
// Pipeline:
//   K0  fill workspace (encoded -inf for segment-max, zeros for sums)
//   K1  fold W_src/W_dst with attn vectors -> WfT [16 x 256] (f16, B-friendly
//       transposed layout: WfT[col*256 + k])
//   K2  node logits: X[N,256] @ Wfold[256,16] via v_wmma_f32_16x16x32_f16
//        cols 0-3  = el_sp (h_sent),  4-7  = er_sp (rel_sp),
//        cols 8-11 = el_ps (rel_ps), 12-15 = er_ps (h_sent)
//   K3  edge passes x3 (max, exp-sum, normalize+accumulate) per edge type
//   K4  finalize: out = h * mean_h(a_sum) + mean_h(bias)

typedef __attribute__((ext_vector_type(16))) _Float16 v16h;
typedef __attribute__((ext_vector_type(8)))  float    v8f;

#define D_FEAT 256
#define H_HEADS 4

// ---- order-preserving float<->uint encoding for atomic max ----
__device__ __forceinline__ unsigned enc_f32(float f) {
  unsigned b = __float_as_uint(f);
  return (b & 0x80000000u) ? ~b : (b | 0x80000000u);
}
__device__ __forceinline__ float dec_f32(unsigned u) {
  return (u & 0x80000000u) ? __uint_as_float(u ^ 0x80000000u)
                           : __uint_as_float(~u);
}
#define ENC_NEG_INF 0x007FFFFFu  // enc_f32(-inf)

__global__ void fill_u32_kernel(unsigned* __restrict__ p, long n, unsigned v) {
  long i = (long)blockIdx.x * blockDim.x + threadIdx.x;
  if (i < n) p[i] = v;
}

__global__ void bias_mean_kernel(const float* __restrict__ bias_pair,
                                 const float* __restrict__ bias_sent,
                                 float* __restrict__ bm_pair,
                                 float* __restrict__ bm_sent) {
  int i = blockIdx.x * blockDim.x + threadIdx.x;  // 0..511
  const float* b = (i < D_FEAT) ? bias_pair : bias_sent;
  float*       o = (i < D_FEAT) ? bm_pair   : bm_sent;
  int d = i & (D_FEAT - 1);
  float acc = 0.f;
#pragma unroll
  for (int h = 0; h < H_HEADS; ++h) acc += b[h * D_FEAT + d];
  o[d] = acc * (1.f / H_HEADS);
}

// WfT[col * 256 + k] (transposed so a B-fragment lane reads 16 contiguous f16):
//  col 0-3  : sum_d W_src[k, h*D+d] * attn_l_sp[h,d]   (h = col & 3)
//  col 4-7  : sum_d W_dst[k, h*D+d] * attn_r_sp[h,d]
//  col 8-11 : sum_d W_src[k, h*D+d] * attn_l_ps[h,d]
//  col 12-15: sum_d W_dst[k, h*D+d] * attn_r_ps[h,d]
__global__ void fold_weights_kernel(const float* __restrict__ W_src,
                                    const float* __restrict__ W_dst,
                                    const float* __restrict__ al_sp,
                                    const float* __restrict__ ar_sp,
                                    const float* __restrict__ al_ps,
                                    const float* __restrict__ ar_ps,
                                    _Float16* __restrict__ WfT) {
  int t = blockIdx.x * blockDim.x + threadIdx.x;  // 0..4095
  int k = t >> 4, col = t & 15, h = col & 3;
  const float* W = (col < 4 || (col >= 8 && col < 12)) ? W_src : W_dst;
  const float* a = (col < 4) ? al_sp : (col < 8) ? ar_sp
                  : (col < 12) ? al_ps : ar_ps;
  const float* wrow = W + (long)k * (H_HEADS * D_FEAT) + h * D_FEAT;
  const float* arow = a + h * D_FEAT;
  float acc = 0.f;
  for (int d = 0; d < D_FEAT; ++d) acc += wrow[d] * arow[d];
  WfT[col * D_FEAT + k] = (_Float16)acc;
}

// Skinny GEMM out[N,16] = X[N,256] @ Wfold[256,16] using WMMA f16->f32.
// 8 waves per block, one 16-row tile per wave. Tail rows are handled by
// CLAMPING the source row (A row m only feeds D row m, and out-of-range D
// rows are never stored) -> zero predication, all loads unconditional/vector.
__global__ __launch_bounds__(256)
void node_logits_kernel(const float* __restrict__ X, int nrows,
                        const _Float16* __restrict__ WfT,
                        float* __restrict__ out) {
  const int wave = threadIdx.x >> 5;
  const int lane = threadIdx.x & 31;
  const int g    = lane >> 4;    // half-wave group
  const int ml   = lane & 15;    // A/D row index (also B/D column)
  const long row0 = ((long)blockIdx.x * 8 + wave) * 16;

  long arow = row0 + ml;
  if (arow >= nrows) arow = nrows - 1;  // clamp: garbage rows never stored
  const float4* xrow = (const float4*)(X + arow * D_FEAT);
  const _Float16* bcol = WfT + ml * D_FEAT;  // this lane's B column

  v8f acc = {};
#pragma unroll
  for (int k0 = 0; k0 < D_FEAT; k0 += 32) {
    // A fragment (16-bit A 16x32): this code's k order gives each lane one contiguous
    // 16-float run: [k0+16g .. +15]  -> 4x b128 loads.
    const int q0 = (k0 + 16 * g) >> 2;       // float4 index of first half
    const int q1 = q0 + 2;                   // float4 index of second half
    float4 c0 = xrow[q0];
    float4 c1 = xrow[q0 + 1];
    float4 c2 = xrow[q1];
    float4 c3 = xrow[q1 + 1];
    v16h a;
    a[0]  = (_Float16)c0.x;  a[1]  = (_Float16)c0.y;
    a[2]  = (_Float16)c0.z;  a[3]  = (_Float16)c0.w;
    a[4]  = (_Float16)c1.x;  a[5]  = (_Float16)c1.y;
    a[6]  = (_Float16)c1.z;  a[7]  = (_Float16)c1.w;
    a[8]  = (_Float16)c2.x;  a[9]  = (_Float16)c2.y;
    a[10] = (_Float16)c2.z;  a[11] = (_Float16)c2.w;
    a[12] = (_Float16)c3.x;  a[13] = (_Float16)c3.y;
    a[14] = (_Float16)c3.z;  a[15] = (_Float16)c3.w;
    // B fragment (32x16): lane ml holds column ml, K = k0+16g .. k0+16g+15,
    // which is 16 contiguous f16 in the transposed WfT -> one 32B load.
    v16h b = *(const v16h*)(bcol + k0 + 16 * g);
    acc = __builtin_amdgcn_wmma_f32_16x16x32_f16(
        /*neg_a=*/false, a, /*neg_b=*/false, b,
        /*c_mod=*/(short)0, acc, /*reuse_a=*/false, /*reuse_b=*/false);
  }
  // D: 32-bit C/D 16x16 layout: VGPR r -> row r + 8*g, col = ml
#pragma unroll
  for (int r = 0; r < 8; ++r) {
    long row = row0 + r + 8 * g;
    if (row < nrows) out[row * 16 + ml] = acc[r];
  }
}

// phase 0: segmented max (encoded-uint atomic max)
// phase 1: segmented sum of exp(e - m)
// phase 2: a = exp(e - m)/s ; atomically accumulate a_sum per dst node
__global__ void edge_pass_kernel(const int* __restrict__ src,
                                 const int* __restrict__ dst, int E,
                                 const float* __restrict__ outSrc, int colSrc,
                                 const float* __restrict__ outDst, int colDst,
                                 unsigned* __restrict__ mEnc,
                                 float* __restrict__ s,
                                 float* __restrict__ asum, int phase) {
  int e = blockIdx.x * blockDim.x + threadIdx.x;
  if (e >= E) return;
  const int u = src[e], v = dst[e];
  const float4 lv = *(const float4*)(outSrc + (long)u * 16 + colSrc);
  const float4 rv = *(const float4*)(outDst + (long)v * 16 + colDst);
  float logit[H_HEADS] = {lv.x * rv.x, lv.y * rv.y, lv.z * rv.z, lv.w * rv.w};
#pragma unroll
  for (int h = 0; h < H_HEADS; ++h) {
    long idx = (long)v * H_HEADS + h;
    if (phase == 0) {
      atomicMax(&mEnc[idx], enc_f32(logit[h]));
    } else {
      float m  = dec_f32(mEnc[idx]);
      float ex = __expf(logit[h] - m);
      if (phase == 1) atomicAdd(&s[idx], ex);
      else            atomicAdd(&asum[idx], ex / s[idx]);
    }
  }
}

// out[node,d] = h[node,d] * mean_h(a_sum[node,h]) + bias_mean[d]
__global__ void finalize_kernel(const float* __restrict__ hfeat,
                                const float* __restrict__ asum,
                                const float* __restrict__ bm,
                                float* __restrict__ outp) {
  long node = blockIdx.x;
  int  d    = threadIdx.x;
  const float4 a = *(const float4*)(asum + node * H_HEADS);
  float am = 0.25f * (a.x + a.y + a.z + a.w);
  long i = node * D_FEAT + d;
  outp[i] = hfeat[i] * am + bm[d];
}

extern "C" void kernel_launch(void* const* d_in, const int* in_sizes, int n_in,
                              void* d_out, int out_size, void* d_ws, size_t ws_size,
                              hipStream_t stream) {
  const float* h_sent    = (const float*)d_in[0];
  const float* h_pair    = (const float*)d_in[1];
  const float* rel_sp    = (const float*)d_in[2];
  const float* rel_ps    = (const float*)d_in[3];
  const float* W_src     = (const float*)d_in[4];
  const float* W_dst     = (const float*)d_in[5];
  const float* attn_l_ps = (const float*)d_in[6];
  const float* attn_r_ps = (const float*)d_in[7];
  const float* attn_l_sp = (const float*)d_in[8];
  const float* attn_r_sp = (const float*)d_in[9];
  const float* bias_sent = (const float*)d_in[10];
  const float* bias_pair = (const float*)d_in[11];
  const int* src_sp = (const int*)d_in[12];
  const int* dst_sp = (const int*)d_in[13];
  const int* src_ps = (const int*)d_in[14];
  const int* dst_ps = (const int*)d_in[15];

  const int NS = in_sizes[0] / D_FEAT;
  const int NP = in_sizes[1] / D_FEAT;
  const int E  = in_sizes[12];

  // ---- carve workspace (~20 MB) ----
  char* ws = (char*)d_ws;
  size_t off = 0;
  auto carve = [&](size_t bytes) -> void* {
    void* p = ws + off;
    off = (off + bytes + 255) & ~(size_t)255;
    return p;
  };
  _Float16* WfT        = (_Float16*)carve(256 * 16 * sizeof(_Float16));
  float*    out_sent   = (float*)carve((size_t)NS * 16 * sizeof(float));
  float*    out_relsp  = (float*)carve((size_t)NP * 16 * sizeof(float));
  float*    out_relps  = (float*)carve((size_t)NP * 16 * sizeof(float));
  unsigned* m_pair     = (unsigned*)carve((size_t)NP * H_HEADS * sizeof(unsigned));
  float*    s_pair     = (float*)carve((size_t)NP * H_HEADS * sizeof(float));
  float*    asum_pair  = (float*)carve((size_t)NP * H_HEADS * sizeof(float));
  unsigned* m_sent     = (unsigned*)carve((size_t)NS * H_HEADS * sizeof(unsigned));
  float*    s_sent     = (float*)carve((size_t)NS * H_HEADS * sizeof(float));
  float*    asum_sent  = (float*)carve((size_t)NS * H_HEADS * sizeof(float));
  float*    bm_pair    = (float*)carve(D_FEAT * sizeof(float));
  float*    bm_sent    = (float*)carve(D_FEAT * sizeof(float));

  float* h_pair_out = (float*)d_out;                       // [NP, 256]
  float* h_sent_out = (float*)d_out + (size_t)NP * D_FEAT; // [NS, 256]

  const int BT = 256;
  auto blks = [](long n, int bt) { return (int)((n + bt - 1) / bt); };

  // K0: init softmax state (every call — ws is not re-poisoned by harness)
  long nP = (long)NP * H_HEADS, nSn = (long)NS * H_HEADS;
  fill_u32_kernel<<<blks(nP, BT),  BT, 0, stream>>>(m_pair, nP, ENC_NEG_INF);
  fill_u32_kernel<<<blks(nSn, BT), BT, 0, stream>>>(m_sent, nSn, ENC_NEG_INF);
  fill_u32_kernel<<<blks(nP, BT),  BT, 0, stream>>>((unsigned*)s_pair,    nP,  0u);
  fill_u32_kernel<<<blks(nP, BT),  BT, 0, stream>>>((unsigned*)asum_pair, nP,  0u);
  fill_u32_kernel<<<blks(nSn, BT), BT, 0, stream>>>((unsigned*)s_sent,    nSn, 0u);
  fill_u32_kernel<<<blks(nSn, BT), BT, 0, stream>>>((unsigned*)asum_sent, nSn, 0u);
  bias_mean_kernel<<<2, BT, 0, stream>>>(bias_pair, bias_sent, bm_pair, bm_sent);

  // K1: fold weights -> WfT [16 x 256] f16 (transposed)
  fold_weights_kernel<<<16, BT, 0, stream>>>(W_src, W_dst, attn_l_sp, attn_r_sp,
                                             attn_l_ps, attn_r_ps, WfT);

  // K2: node logits via WMMA (8 waves/block -> 128 rows/block)
  auto gemm_grid = [](int n) { return (((n + 15) / 16) + 7) / 8; };
  node_logits_kernel<<<gemm_grid(NS), BT, 0, stream>>>(h_sent, NS, WfT, out_sent);
  node_logits_kernel<<<gemm_grid(NP), BT, 0, stream>>>(rel_sp, NP, WfT, out_relsp);
  node_logits_kernel<<<gemm_grid(NP), BT, 0, stream>>>(rel_ps, NP, WfT, out_relps);

  // K3: segmented softmax + a_sum accumulation (3 passes per edge type)
  int eg = blks(E, BT);
  for (int phase = 0; phase < 3; ++phase) {
    // sp: src in h_sent (cols 0-3), dst in rel_sp (cols 4-7), segments = pairs
    edge_pass_kernel<<<eg, BT, 0, stream>>>(src_sp, dst_sp, E,
                                            out_sent, 0, out_relsp, 4,
                                            m_pair, s_pair, asum_pair, phase);
    // ps: src in rel_ps (cols 8-11), dst in h_sent (cols 12-15), segments = sents
    edge_pass_kernel<<<eg, BT, 0, stream>>>(src_ps, dst_ps, E,
                                            out_relps, 8, out_sent, 12,
                                            m_sent, s_sent, asum_sent, phase);
  }

  // K4: finalize both outputs (tuple order: h_pair_out, then h_sent_out)
  finalize_kernel<<<NP, D_FEAT, 0, stream>>>(h_pair, asum_pair, bm_pair, h_pair_out);
  finalize_kernel<<<NS, D_FEAT, 0, stream>>>(h_sent, asum_sent, bm_sent, h_sent_out);
}
